// EEGGraphConvNet_35734127903265
// MI455X (gfx1250) — matrix operand order repaired and checked by reading, not compiled, not run on hardware
//
#include <hip/hip_runtime.h>
#include <hip/hip_bf16.h>

typedef __attribute__((ext_vector_type(16))) _Float16 v16h;
typedef __attribute__((ext_vector_type(8)))  _Float16 v8h;
typedef __attribute__((ext_vector_type(8)))  float    v8f;

#define NUMG   50000          // graphs (known from reference)
#define NPG    8              // nodes per graph
#define SP     80             // row-major activation LDS stride in halves (160B, 16B-aligned)
#define TR     40             // transposed tile column stride in halves (80B, 16B-aligned)
#define SLOPE  0.01f
#define EPS    1e-5f

__device__ __forceinline__ float lrelu(float x){ return x >= 0.f ? x : SLOPE * x; }
__device__ __forceinline__ int   imin(int a, int b){ return a < b ? a : b; }

__device__ __forceinline__ v8f wmma_f16(v16h a, v16h b, v8f c){
  // D(16x16,f32) = A(16x32,f16) * B(32x16,f16) + C
  return __builtin_amdgcn_wmma_f32_16x16x32_f16(false, a, false, b, (short)0, c, false, false);
}

// ---- fragment builders: all loads UNCONDITIONAL (padding is zero data) ---

// A (16x32 f16) from row-major f16 LDS activations, row = lane%16.
// lane<16: K = {k0..k0+7, k0+16..k0+23}; lane>=16: +8. Two contiguous 16B chunks.
__device__ __forceinline__ v16h load_a_stage(const _Float16* S, int ld, int k0){
  int lane = threadIdx.x & 31;
  int hi   = lane >> 4;
  const _Float16* p = S + (lane & 15) * ld + k0 + 8 * hi;
  v8h lo = *(const v8h*)(p);
  v8h hh = *(const v8h*)(p + 16);
  v16h a;
#pragma unroll
  for (int i = 0; i < 8; ++i){ a[i] = lo[i]; a[i + 8] = hh[i]; }
  return a;
}

// B (32x16 f16) from LDS-staged transposed weights Wl[n*KP + k] (zero padded).
__device__ __forceinline__ v16h load_b_stage(const _Float16* Wl, int KP, int k0, int n0){
  int lane = threadIdx.x & 31;
  int hi   = lane >> 4;
  const _Float16* p = Wl + (n0 + (lane & 15)) * KP + k0 + 8 * hi;
  v8h lo = *(const v8h*)(p);
  v8h hh = *(const v8h*)(p + 16);
  v16h b;
#pragma unroll
  for (int i = 0; i < 8; ++i){ b[i] = lo[i]; b[i + 8] = hh[i]; }
  return b;
}

// B (32x16 f16) from a TRANSPOSED [col][row] f16 LDS tile (rows 16..31 pre-zeroed).
// lane holds col n = n0+lane%16; K = {8*hi+0..7, 16+8*hi+0..7} -> 2x ds_load_b128.
__device__ __forceinline__ v16h load_bT_T(const _Float16* St, int n0){
  int lane = threadIdx.x & 31;
  int hi   = lane >> 4;
  const _Float16* p = St + (n0 + (lane & 15)) * TR + 8 * hi;
  v8h lo = *(const v8h*)(p);
  v8h hh = *(const v8h*)(p + 16);
  v16h b;
#pragma unroll
  for (int i = 0; i < 8; ++i){ b[i] = lo[i]; b[i + 8] = hh[i]; }
  return b;
}

// Block-diagonal adjacency (2 graphs) as A-frag; clamped loads + cndmask select.
__device__ __forceinline__ v16h load_adj_frag(const float* __restrict__ ew, int gA){
  int lane = threadIdx.x & 31;
  int r    = lane & 15;
  int hi   = lane >> 4;
  int bsel = r >> 3;
  int i    = r & 7;
  int g    = gA + bsel;
  bool act = (hi == bsel) && (g < NUMG);
  const float* base = ew + (long)imin(g, NUMG - 1) * 64 + i;
  v16h a;
#pragma unroll
  for (int h = 0; h < 8; ++h){
    float v = base[h * 8];                 // unconditional, in-bounds
    a[h]     = (_Float16)(act ? v : 0.f);  // cndmask
    a[h + 8] = (_Float16)0.f;              // K 16..31 padding
  }
  return a;
}

// Stage a weight matrix W[K][N] (f32, global) into LDS transposed f16 [NR][KP], zero padded.
__device__ __forceinline__ void stage_w(_Float16* dst, const float* __restrict__ W,
                                        int K, int N, int KP, int NR){
  for (int idx = threadIdx.x; idx < NR * KP; idx += 256){
    int n = idx / KP, k = idx - n * KP;
    float v = (k < K && n < N) ? W[k * N + n] : 0.f;
    dst[idx] = (_Float16)v;
  }
}

// ---- layer helpers -------------------------------------------------------

// Transform: row-major Sin -> WMMA -> TRANSPOSED St ([col][row], packed b128 store)
template<int FIN, int FOUT>
__device__ __forceinline__ void gcn_transform(const _Float16* Sin, _Float16* St,
                                              const _Float16* Wl){
  constexpr int KT = (FIN + 31) / 32;
  constexpr int NT = (FOUT + 15) / 16;
  constexpr int KP = 32 * KT + 8;
  int lane = threadIdx.x & 31, hi = lane >> 4;
#pragma unroll
  for (int nt = 0; nt < NT; ++nt){
    v8f acc = {};
#pragma unroll
    for (int kt = 0; kt < KT; ++kt){
      v16h a = load_a_stage(Sin, SP, kt * 32);
      v16h b = load_b_stage(Wl, KP, kt * 32, nt * 16);
      acc = wmma_f16(a, b, acc);
    }
    int c = nt * 16 + (lane & 15);
    v8h ph;                                 // rows 8*hi .. 8*hi+7 of column c
#pragma unroll
    for (int i = 0; i < 8; ++i) ph[i] = (_Float16)acc[i];
    *(v8h*)(St + c * TR + 8 * hi) = ph;     // single ds_store_b128
  }
}

// Aggregate: Adj (A) x St-transposed (B) -> bias + lrelu -> row-major Sout
template<int FOUT>
__device__ __forceinline__ void gcn_aggregate(const _Float16* St, _Float16* Sout, v16h adj,
                                              const float* __restrict__ bias){
  constexpr int NT = (FOUT + 15) / 16;
  int lane = threadIdx.x & 31, hi = lane >> 4;
#pragma unroll
  for (int nt = 0; nt < NT; ++nt){
    v16h b = load_bT_T(St, nt * 16);
    v8f d = {};
    d = wmma_f16(adj, b, d);
    int c = nt * 16 + (lane & 15);          // c < FOUT always (FOUT multiple of 16)
    float bb = bias[c];
#pragma unroll
    for (int i = 0; i < 8; ++i)
      Sout[(i + 8 * hi) * SP + c] = (_Float16)lrelu(d[i] + bb);
  }
}

template<int FIN, int FOUT>
__device__ __forceinline__ void fc_layer(const _Float16* Sin, int ldin, _Float16* Sout, int ldout,
                                         const _Float16* Wl, const float* __restrict__ bias){
  constexpr int KT = (FIN + 31) / 32;
  constexpr int NT = (FOUT + 15) / 16;
  constexpr int KP = 32 * KT + 8;
  int lane = threadIdx.x & 31, hi = lane >> 4;
#pragma unroll
  for (int nt = 0; nt < NT; ++nt){
    v8f acc = {};
#pragma unroll
    for (int kt = 0; kt < KT; ++kt){
      v16h a = load_a_stage(Sin, ldin, kt * 32);
      v16h b = load_b_stage(Wl, KP, kt * 32, nt * 16);
      acc = wmma_f16(a, b, acc);
    }
    int c  = nt * 16 + (lane & 15);
    float bv = bias[imin(c, FOUT - 1)];     // unconditional clamped load
    float bb = (c < FOUT) ? bv : 0.f;       // cndmask -> padded cols become exactly 0
#pragma unroll
    for (int i = 0; i < 8; ++i)
      Sout[(i + 8 * hi) * ldout + c] = (_Float16)lrelu(acc[i] + bb);
  }
}

// ---- kernel 0: zero BN accumulators -------------------------------------
__global__ void init_kernel(float* accum){
  if (threadIdx.x < 128) accum[threadIdx.x] = 0.f;
}

// ---- kernel 1: fused GCN x4 (tile-local), spill h4 f16, BN stats --------
__global__ __launch_bounds__(256, 1)
void gcn_fused_kernel(const float* __restrict__ x, const float* __restrict__ ew,
                      const float* __restrict__ W1, const float* __restrict__ b1,
                      const float* __restrict__ W2, const float* __restrict__ b2,
                      const float* __restrict__ W3, const float* __restrict__ b3,
                      const float* __restrict__ W4, const float* __restrict__ b4,
                      _Float16* __restrict__ h4, float* __restrict__ accum)
{
  // staged weights (transposed f16, zero padded): W1@0(16x40) W2@640(32x40) W3@1920(64x40) W4@4480(64x72)
  __shared__ _Float16 Wst[9088];
  __shared__ _Float16 SinA[8 * 16 * SP];    // per-wave row-major input tile (f16)
  __shared__ _Float16 StA [8 * 64 * TR];    // per-wave TRANSPOSED transform tile [col][row]
  __shared__ float sAcc[128];

  int tid = threadIdx.x, wave = tid >> 5, lane = tid & 31, hi = lane >> 4;

  stage_w(Wst + 0,    W1,  6, 16, 40, 16);
  stage_w(Wst + 640,  W2, 16, 32, 40, 32);
  stage_w(Wst + 1920, W3, 32, 64, 40, 64);
  stage_w(Wst + 4480, W4, 64, 50, 72, 64);
  if (tid < 128) sAcc[tid] = 0.f;
  __syncthreads();

  _Float16* Sin = SinA + wave * (16 * SP);
  _Float16* St  = StA  + wave * (64 * TR);

  // zero St rows 16..31 of every column once (K-padding for aggregation B frags)
  for (int idx = lane; idx < 64 * 8; idx += 32){     // 16 uniform iterations (dwords)
    int col = idx >> 3, d = idx & 7;
    ((float*)(St + col * TR + 16))[d] = 0.f;
  }

  for (int t = 0; t < 2; ++t){                        // 2 tiles per wave
    int tile = wave * 2 + t;
    int gA   = blockIdx.x * 32 + tile * 2;            // 32 graphs / block

    // zero Sin cols 0..31 (as dwords), then overlay x into cols 0..5
    for (int idx = lane; idx < 16 * 16; idx += 32){
      int r = idx >> 4, c2 = idx & 15;
      ((float*)(Sin + r * SP))[c2] = 0.f;
    }
    for (int idx = lane; idx < 16 * 6; idx += 32){    // 3 uniform iterations
      int r = idx / 6, c = idx - r * 6;
      int g = gA + (r >> 3);
      float v = x[((long)imin(g, NUMG - 1) * NPG + (r & 7)) * 6 + c];
      Sin[r * SP + c] = (_Float16)(g < NUMG ? v : 0.f);
    }

    v16h adj = load_adj_frag(ew, gA);                 // reused across all 4 layers

    gcn_transform<6, 16>(Sin, St, Wst + 0);
    gcn_aggregate<16>(St, Sin, adj, b1);
    gcn_transform<16, 32>(Sin, St, Wst + 640);
    gcn_aggregate<32>(St, Sin, adj, b2);
    gcn_transform<32, 64>(Sin, St, Wst + 1920);
    gcn_aggregate<64>(St, Sin, adj, b3);
    gcn_transform<64, 50>(Sin, St, Wst + 4480);

    // layer-4 aggregate (no activation): spill h4 (f16) + BN sum/sumsq
#pragma unroll
    for (int nt = 0; nt < 4; ++nt){
      v16h b = load_bT_T(St, nt * 16);
      v8f d = {};
      d = wmma_f16(adj, b, d);
      int c = nt * 16 + (lane & 15);
      int g = gA + hi;                                // lanes<16: gA, lanes>=16: gA+1
      float bb = b4[imin(c, 49)];                     // unconditional clamped load
      bool valid = (g < NUMG) && (c < 50);
      if (valid){                                     // single divergence point
        float sum = 0.f, sq = 0.f;
        _Float16* dst = h4 + ((long)g * NPG) * 50 + c;
#pragma unroll
        for (int i = 0; i < 8; ++i){
          float v = d[i] + bb;
          dst[i * 50] = (_Float16)v;
          sum += v; sq += v * v;
        }
        atomicAdd(&sAcc[c],      sum);
        atomicAdd(&sAcc[64 + c], sq);
      }
    }
  }

  __syncthreads();
  if (tid < 50){
    atomicAdd(&accum[tid],      sAcc[tid]);
    atomicAdd(&accum[64 + tid], sAcc[64 + tid]);
  }
}

// ---- kernel 2: fold BN stats into scale/shift ---------------------------
__global__ void bn_kernel(const float* __restrict__ accum,
                          const float* __restrict__ gamma, const float* __restrict__ beta,
                          float* __restrict__ bnp){
  int c = threadIdx.x;
  if (c < 50){
    const float invN = 1.0f / (float)(NUMG * NPG);
    float mu  = accum[c] * invN;
    float var = accum[64 + c] * invN - mu * mu;
    float a   = gamma[c] * rsqrtf(var + EPS);
    bnp[c]      = a;
    bnp[64 + c] = beta[c] - mu * a;
  }
}

// ---- kernel 3: BN + lrelu + pool + FC(50->30->20->2) --------------------
__global__ __launch_bounds__(256, 1)
void head_kernel(const _Float16* __restrict__ h4, const float* __restrict__ bnp,
                 const float* __restrict__ Wf1, const float* __restrict__ bf1,
                 const float* __restrict__ Wf2, const float* __restrict__ bf2,
                 const float* __restrict__ Wf3, const float* __restrict__ bf3,
                 float* __restrict__ out)
{
  // staged weights: Wf1@0(32x72) Wf2@2304(32x40) Wf3@3584(16x40)
  __shared__ _Float16 Wst[4224];
  __shared__ _Float16 S0A[8 * 16 * SP];
  __shared__ _Float16 S1A[8 * 16 * 48];
  int tid = threadIdx.x, wave = tid >> 5, lane = tid & 31, hi = lane >> 4;

  stage_w(Wst + 0,    Wf1, 50, 30, 72, 32);
  stage_w(Wst + 2304, Wf2, 30, 20, 40, 32);
  stage_w(Wst + 3584, Wf3, 20,  2, 40, 16);
  __syncthreads();

  _Float16* S0 = S0A + wave * (16 * SP);
  _Float16* S1 = S1A + wave * (16 * 48);
  int gt = blockIdx.x * 128 + wave * 16;              // this wave's 16 graphs

  // zero S0 cols 50..63 (pad for K up to 64)
  for (int idx = lane; idx < 16 * 14; idx += 32){
    int gl = idx / 14, c = 50 + idx - (idx / 14) * 14;
    S0[gl * SP + c] = (_Float16)0.f;
  }

  // BN + lrelu + global_add_pool -> S0[16 graphs][50] (f16)
  for (int idx = lane; idx < 16 * 50; idx += 32){     // exactly 25 uniform iterations
    int gl = idx / 50, c = idx - gl * 50;
    int g  = gt + gl;
    float a = bnp[c], bs = bnp[64 + c];
    const _Float16* p = h4 + (long)imin(g, NUMG - 1) * NPG * 50 + c;
    float s = 0.f;
#pragma unroll
    for (int n = 0; n < NPG; ++n)
      s += lrelu(a * (float)p[n * 50] + bs);
    S0[gl * SP + c] = (_Float16)(g < NUMG ? s : 0.f);
  }

  fc_layer<50, 30>(S0, SP, S1, 48, Wst + 0,    bf1);  // cols 30,31 of S1 become exact 0
  fc_layer<30, 20>(S1, 48, S0, SP, Wst + 2304, bf2);  // cols 20..31 of S0 become exact 0

  // final 20 -> 2
  v16h a = load_a_stage(S0, SP, 0);
  v16h b = load_b_stage(Wst + 3584, 40, 0, 0);
  v8f d = {};
  d = wmma_f16(a, b, d);
  int c = lane & 15;
  if (c < 2){
    float bb = bf3[c];
#pragma unroll
    for (int i = 0; i < 8; ++i){
      int g = gt + i + 8 * hi;
      if (g < NUMG) out[(long)g * 2 + c] = d[i] + bb;
    }
  }
}

// ---- launch --------------------------------------------------------------
extern "C" void kernel_launch(void* const* d_in, const int* in_sizes, int n_in,
                              void* d_out, int out_size, void* d_ws, size_t ws_size,
                              hipStream_t stream)
{
  (void)in_sizes; (void)n_in; (void)out_size; (void)ws_size;

  const float* x     = (const float*)d_in[0];
  const float* ew    = (const float*)d_in[2];
  const float* W1    = (const float*)d_in[5];
  const float* b1    = (const float*)d_in[6];
  const float* W2    = (const float*)d_in[7];
  const float* b2    = (const float*)d_in[8];
  const float* W3    = (const float*)d_in[9];
  const float* b3    = (const float*)d_in[10];
  const float* W4    = (const float*)d_in[11];
  const float* b4    = (const float*)d_in[12];
  const float* gamma = (const float*)d_in[13];
  const float* beta  = (const float*)d_in[14];
  const float* Wf1   = (const float*)d_in[15];
  const float* bf1   = (const float*)d_in[16];
  const float* Wf2   = (const float*)d_in[17];
  const float* bf2   = (const float*)d_in[18];
  const float* Wf3   = (const float*)d_in[19];
  const float* bf3   = (const float*)d_in[20];
  float* out = (float*)d_out;

  // workspace: h4 (f16, 40MB) | accum[128] | bnp[128]
  char* ws = (char*)d_ws;
  _Float16* h4 = (_Float16*)ws;
  float* accum = (float*)(ws + 40000000);
  float* bnp   = (float*)(ws + 40000512);

  init_kernel<<<1, 128, 0, stream>>>(accum);
  gcn_fused_kernel<<<(NUMG + 31) / 32, 256, 0, stream>>>(
      x, ew, W1, b1, W2, b2, W3, b3, W4, b4, h4, accum);
  bn_kernel<<<1, 64, 0, stream>>>(accum, gamma, beta, bnp);
  head_kernel<<<(NUMG + 127) / 128, 256, 0, stream>>>(
      h4, bnp, Wf1, bf1, Wf2, bf2, Wf3, bf3, out);
}
